// DecoderLayer_15642270892252
// MI455X (gfx1250) — compile-verified, matrix operand order checked by reading, not run on hardware
//
#include <hip/hip_runtime.h>
#include <hip/hip_bf16.h>
#include <stdint.h>

#define B_   2
#define L_   2048
#define D_   1024
#define H_   16
#define DFF_ 4096
#define DH_  64
#define M_   (B_ * L_)        // 4096 rows total
#define EPSL 1e-6f

typedef __attribute__((ext_vector_type(16))) __bf16 v16bf;
typedef __attribute__((ext_vector_type(8)))  float  v8f;
typedef __attribute__((ext_vector_type(4)))  int    v4i;
typedef unsigned short u16t;

#if defined(__AMDGCN__) &&                                            \
    __has_builtin(__builtin_amdgcn_global_load_async_to_lds_b128) &&  \
    __has_builtin(__builtin_amdgcn_s_wait_asynccnt)
#define USE_ASYNC_LDS 1
typedef v4i __attribute__((address_space(1)))* gv4p;   // global int4*
typedef v4i __attribute__((address_space(3)))* lv4p;   // LDS int4* (32-bit)
#else
#define USE_ASYNC_LDS 0
#endif

union FragBF { v16bf v; u16t u[16]; uint4 q[2]; };

__device__ __forceinline__ u16t f2bf(float f) {
  uint32_t u = __float_as_uint(f);
  uint32_t r = 0x7FFFu + ((u >> 16) & 1u);
  return (u16t)((u + r) >> 16);
}

// Load a 16x32 bf16 fragment in WMMA A/B layout from a row-major matrix.
// Lane (0..15):  row = row0+lane,      holds K = k0+[0..7] and k0+[16..23]
// Lane (16..31): row = row0+lane-16,   holds K = k0+[8..15] and k0+[24..31]
__device__ __forceinline__ v16bf load_frag_g(const u16t* __restrict__ base,
                                             int ld, long row0, int k0) {
  int lane = threadIdx.x & 31;
  int idx  = lane & 15;
  int half = lane >> 4;
  const u16t* p = base + (size_t)(row0 + idx) * ld + k0 + half * 8;
  FragBF f;
  f.q[0] = *(const uint4*)p;          // K chunk 0
  f.q[1] = *(const uint4*)(p + 16);   // K chunk +16
  return f.v;
}

// Same fragment layout, loading from an LDS tile (ld = padded row length).
__device__ __forceinline__ v16bf load_frag_lds(const u16t* base, int ld,
                                               int row0) {
  int lane = threadIdx.x & 31;
  int idx  = lane & 15;
  int half = lane >> 4;
  const u16t* p = base + (row0 + idx) * ld + half * 8;
  FragBF f;
  f.q[0] = *(const uint4*)p;
  f.q[1] = *(const uint4*)(p + 16);
  return f.v;
}

__device__ __forceinline__ v8f wmma_bf16(v16bf a, v16bf b, v8f c) {
  return __builtin_amdgcn_wmma_f32_16x16x32_bf16(false, a, false, b,
                                                 (short)0, c, false, false);
}

// ---------------------------------------------------------------------------
// Elementwise f32 -> bf16 cast
// ---------------------------------------------------------------------------
__global__ void cast_bf16_kernel(const float* __restrict__ in,
                                 u16t* __restrict__ out, int n) {
  for (int i = blockIdx.x * blockDim.x + threadIdx.x; i < n;
       i += gridDim.x * blockDim.x)
    out[i] = f2bf(in[i]);
}

// ---------------------------------------------------------------------------
// C[M x N] = A[M x K] (bf16) * W[N x K]^T (bf16) + bias, optional relu.
// Block = 256 thr = 8 waves; block tile 128x128; wave tile 32x64 (2x4 frags).
// A/B tiles staged to LDS with async global->LDS copies (double buffered),
// fragment reads then come from LDS (ds_load_b128).
// ---------------------------------------------------------------------------
#define KSTEP 32
#define LPAD  40   // 32 bf16 + 8 pad (80 B rows, 16B aligned)

__global__ __launch_bounds__(256)
void gemm_bf16_kernel(const u16t* __restrict__ A, int lda,
                      const u16t* __restrict__ W, int ldw,
                      const float* __restrict__ bias,
                      float* __restrict__ outF, u16t* __restrict__ outB,
                      int N, int K, int relu) {
  __shared__ u16t As[2][128][LPAD];
  __shared__ u16t Bs[2][128][LPAD];

  const int t    = threadIdx.x;
  const int lane = t & 31;
  const int wid  = t >> 5;
  const int rowb = blockIdx.y * 128;             // block row base (M)
  const int colb = blockIdx.x * 128;             // block col base (N)
  const int wrow = (wid & 3) * 32;               // wave row offset in tile
  const int wcol = (wid >> 2) * 64;              // wave col offset in tile

  const v8f z = {0.f, 0.f, 0.f, 0.f, 0.f, 0.f, 0.f, 0.f};
  v8f acc[2][4];
#pragma unroll
  for (int i = 0; i < 2; ++i)
#pragma unroll
    for (int j = 0; j < 4; ++j) acc[i][j] = z;

  // Stage one K-chunk (128 rows x 32 cols of both A and W) into LDS buf.
  // 512 16-byte chunks per matrix; 256 threads -> 2 chunks each.
  auto stage = [&](int buf, int k0) {
#pragma unroll
    for (int c = 0; c < 2; ++c) {
      int ch  = t + c * 256;
      int r   = ch >> 2;
      int col = (ch & 3) * 8;
      const u16t* ga = A + (size_t)(rowb + r) * lda + k0 + col;
      const u16t* gw = W + (size_t)(colb + r) * ldw + k0 + col;
#if USE_ASYNC_LDS
      // AS1 global pointer: same bit pattern as generic.
      // AS3 LDS pointer: low 32 bits of the generic address are the LDS
      // offset (ISA aperture rule).
      __builtin_amdgcn_global_load_async_to_lds_b128(
          (gv4p)(uintptr_t)ga,
          (lv4p)(unsigned int)(uintptr_t)&As[buf][r][col], 0, 0);
      __builtin_amdgcn_global_load_async_to_lds_b128(
          (gv4p)(uintptr_t)gw,
          (lv4p)(unsigned int)(uintptr_t)&Bs[buf][r][col], 0, 0);
#else
      *(uint4*)&As[buf][r][col] = *(const uint4*)ga;
      *(uint4*)&Bs[buf][r][col] = *(const uint4*)gw;
#endif
    }
  };

  auto compute = [&](int buf) {
    v16bf a0 = load_frag_lds(&As[buf][0][0], LPAD, wrow);
    v16bf a1 = load_frag_lds(&As[buf][0][0], LPAD, wrow + 16);
#pragma unroll
    for (int j = 0; j < 4; ++j) {
      v16bf b = load_frag_lds(&Bs[buf][0][0], LPAD, wcol + j * 16);
      acc[0][j] = wmma_bf16(a0, b, acc[0][j]);
      acc[1][j] = wmma_bf16(a1, b, acc[1][j]);
    }
  };

  stage(0, 0);
  int buf = 0;
  for (int k0 = 0; k0 < K; k0 += KSTEP) {
#if USE_ASYNC_LDS
    __builtin_amdgcn_s_wait_asynccnt(0);
#endif
    __syncthreads();
    if (k0 + KSTEP < K) stage(buf ^ 1, k0 + KSTEP);
    compute(buf);
    buf ^= 1;
  }

  const int idx = lane & 15, half = lane >> 4;
#pragma unroll
  for (int i = 0; i < 2; ++i)
#pragma unroll
    for (int j = 0; j < 4; ++j) {
      int c = colb + wcol + j * 16 + idx;
      float bs = bias ? bias[c] : 0.0f;
#pragma unroll
      for (int v = 0; v < 8; ++v) {
        int r = rowb + wrow + i * 16 + v + 8 * half;
        float val = acc[i][j][v] + bs;
        if (relu) val = fmaxf(val, 0.0f);
        if (outF) outF[(size_t)r * N + c] = val;
        if (outB) outB[(size_t)r * N + c] = f2bf(val);
      }
    }
}

// ---------------------------------------------------------------------------
// Flash-style causal attention. One wave (32 thr) per 16-query tile per (b,h).
// qkv layout: [B*L][3*D], q at col h*64, k at D + h*64, v at 2D + h*64.
// Output obuf: bf16 [B*L][D].
// ---------------------------------------------------------------------------
__global__ __launch_bounds__(32)
void attn_kernel(const u16t* __restrict__ qkv, u16t* __restrict__ obuf) {
  __shared__ float S[16][32];
  __shared__ u16t  P[16][32];
  __shared__ u16t  VT[64][32];      // V transposed: VT[dh][key]
  __shared__ float mrow[16], lrow[16], alphas[16];

  const int lane = threadIdx.x;
  const int idx  = lane & 15, half = lane >> 4;
  const int bh   = blockIdx.x;
  const int b    = bh / H_, h = bh % H_;
  const int q0   = blockIdx.y * 16;
  const int ld   = 3 * D_;
  const long baserow = (long)b * L_;
  const float scale = 0.125f;       // 1/sqrt(64)

  if (lane < 16) { mrow[lane] = -__builtin_inff(); lrow[lane] = 0.0f; }
  __syncthreads();

  // Q fragments held across the whole key loop (K-dim = DH = 64 -> 2 frags)
  v16bf qa0 = load_frag_g(qkv, ld, baserow + q0, h * DH_ + 0);
  v16bf qa1 = load_frag_g(qkv, ld, baserow + q0, h * DH_ + 32);

  const v8f z = {0.f, 0.f, 0.f, 0.f, 0.f, 0.f, 0.f, 0.f};
  v8f o[4] = {z, z, z, z};

  const int nblocks = (q0 + 15) / 32 + 1;   // causal: keys up to q0+15
  for (int kb = 0; kb < nblocks; ++kb) {
    const int key0 = kb * 32;

    // ---- scores S = (Q K^T) * scale, masked, into LDS -------------------
#pragma unroll
    for (int nf = 0; nf < 2; ++nf) {
      v16bf kb0 = load_frag_g(qkv, ld, baserow + key0 + nf * 16,
                              D_ + h * DH_ + 0);
      v16bf kb1 = load_frag_g(qkv, ld, baserow + key0 + nf * 16,
                              D_ + h * DH_ + 32);
      v8f s = z;
      s = wmma_bf16(qa0, kb0, s);
      s = wmma_bf16(qa1, kb1, s);
      int kg = key0 + nf * 16 + idx;
#pragma unroll
      for (int v = 0; v < 8; ++v) {
        int m = v + 8 * half;
        S[m][nf * 16 + idx] =
            (kg <= q0 + m) ? s[v] * scale : -__builtin_inff();
      }
    }

    // ---- stage V tile transposed into LDS (lane = one key row) ----------
    {
      const u16t* vrow =
          qkv + (size_t)(baserow + key0 + lane) * ld + 2 * D_ + h * DH_;
#pragma unroll
      for (int d = 0; d < DH_; ++d) VT[d][lane] = vrow[d];
    }
    __syncthreads();

    // ---- online softmax (lanes 0..15, one row each) ----------------------
    if (lane < 16) {
      int r = lane;
      float t = -__builtin_inff();
      for (int j = 0; j < 32; ++j) t = fmaxf(t, S[r][j]);
      float nm  = fmaxf(mrow[r], t);
      float al  = __expf(mrow[r] - nm);
      float sum = 0.0f;
      for (int j = 0; j < 32; ++j) {
        float p = __expf(S[r][j] - nm);
        P[r][j] = f2bf(p);
        sum += p;
      }
      lrow[r]   = al * lrow[r] + sum;
      mrow[r]   = nm;
      alphas[r] = al;
    }
    __syncthreads();

    // ---- rescale accumulators by alpha ----------------------------------
#pragma unroll
    for (int v = 0; v < 8; ++v) {
      float al = alphas[v + 8 * half];
#pragma unroll
      for (int nf = 0; nf < 4; ++nf) o[nf][v] *= al;
    }

    // ---- O += P @ V  (A = P 16x32 bf16 from LDS, B = VT rows) ------------
    FragBF pa;
#pragma unroll
    for (int j = 0; j < 8; ++j) {
      pa.u[j]     = P[idx][half * 8 + j];
      pa.u[8 + j] = P[idx][16 + half * 8 + j];
    }
#pragma unroll
    for (int nf = 0; nf < 4; ++nf) {
      FragBF vb;
#pragma unroll
      for (int j = 0; j < 8; ++j) {
        vb.u[j]     = VT[nf * 16 + idx][half * 8 + j];
        vb.u[8 + j] = VT[nf * 16 + idx][16 + half * 8 + j];
      }
      o[nf] = wmma_bf16(pa.v, vb.v, o[nf]);
    }
    __syncthreads();
  }

  // ---- normalize and store (bf16, layout [B*L][D]) -----------------------
#pragma unroll
  for (int v = 0; v < 8; ++v) {
    float inv = 1.0f / lrow[v + 8 * half];
    long  m   = baserow + q0 + v + 8 * half;
#pragma unroll
    for (int nf = 0; nf < 4; ++nf) {
      int dcol = h * DH_ + nf * 16 + idx;
      obuf[(size_t)m * D_ + dcol] = f2bf(o[nf][v] * inv);
    }
  }
}

// ---------------------------------------------------------------------------
// LayerNorm with fused residual: y = LN(a + b) * gamma + beta
// Matches reference: ddof=1 std, divide by (std + eps).
// One 256-thread block per row (D = 1024 -> 4 elements / thread).
// ---------------------------------------------------------------------------
__global__ __launch_bounds__(256)
void ln_kernel(const float* __restrict__ a, const float* __restrict__ bres,
               const float* __restrict__ gamma, const float* __restrict__ beta,
               float* __restrict__ outF, u16t* __restrict__ outB) {
  __shared__ float red[256];
  const int row = blockIdx.x;
  const int t   = threadIdx.x;
  const float* pa = a + (size_t)row * D_;
  const float* pb = bres ? bres + (size_t)row * D_ : nullptr;

  float vals[4];
  float s = 0.0f;
#pragma unroll
  for (int i = 0; i < 4; ++i) {
    int c = t + i * 256;
    float v = pa[c] + (pb ? pb[c] : 0.0f);
    vals[i] = v;
    s += v;
  }
  red[t] = s; __syncthreads();
  for (int off = 128; off > 0; off >>= 1) {
    if (t < off) red[t] += red[t + off];
    __syncthreads();
  }
  float mean = red[0] / (float)D_;
  __syncthreads();

  float s2 = 0.0f;
#pragma unroll
  for (int i = 0; i < 4; ++i) { float d = vals[i] - mean; s2 += d * d; }
  red[t] = s2; __syncthreads();
  for (int off = 128; off > 0; off >>= 1) {
    if (t < off) red[t] += red[t + off];
    __syncthreads();
  }
  float var   = red[0] / (float)(D_ - 1);
  float denom = sqrtf(var) + EPSL;

#pragma unroll
  for (int i = 0; i < 4; ++i) {
    int c = t + i * 256;
    float y = (vals[i] - mean) / denom * gamma[c] + beta[c];
    if (outF) outF[(size_t)row * D_ + c] = y;
    if (outB) outB[(size_t)row * D_ + c] = f2bf(y);
  }
}

// ---------------------------------------------------------------------------
extern "C" void kernel_launch(void* const* d_in, const int* in_sizes, int n_in,
                              void* d_out, int out_size, void* d_ws,
                              size_t ws_size, hipStream_t stream) {
  // Inputs (setup_inputs order):
  const float* x     = (const float*)d_in[0];
  /* mask d_in[1] ignored: causal mask applied analytically */
  const float* Wqkv  = (const float*)d_in[2];
  const float* bqkv  = (const float*)d_in[3];
  const float* Wo    = (const float*)d_in[4];
  const float* bo    = (const float*)d_in[5];
  const float* ln1_a = (const float*)d_in[6];
  const float* ln1_b = (const float*)d_in[7];
  const float* W1    = (const float*)d_in[8];
  const float* b1    = (const float*)d_in[9];
  const float* W2    = (const float*)d_in[10];
  const float* b2    = (const float*)d_in[11];
  const float* ln2_a = (const float*)d_in[12];
  const float* ln2_b = (const float*)d_in[13];
  float* out = (float*)d_out;

  // Workspace layout
  char* ws = (char*)d_ws;
  size_t off = 0;
  auto alloc = [&](size_t bytes) {
    size_t o = off;
    off += (bytes + 255) & ~(size_t)255;
    return o;
  };
  u16t*  xbf    = (u16t*)(ws + alloc((size_t)M_ * D_ * 2));
  u16t*  wqkvbf = (u16t*)(ws + alloc((size_t)3 * D_ * D_ * 2));
  u16t*  wobf   = (u16t*)(ws + alloc((size_t)D_ * D_ * 2));
  u16t*  w1bf   = (u16t*)(ws + alloc((size_t)DFF_ * D_ * 2));
  u16t*  w2bf   = (u16t*)(ws + alloc((size_t)D_ * DFF_ * 2));
  u16t*  qkvbf  = (u16t*)(ws + alloc((size_t)M_ * 3 * D_ * 2));
  u16t*  obf    = (u16t*)(ws + alloc((size_t)M_ * D_ * 2));
  float* projf  = (float*)(ws + alloc((size_t)M_ * D_ * 4));
  float* hf     = (float*)(ws + alloc((size_t)M_ * D_ * 4));
  u16t*  hbf    = (u16t*)(ws + alloc((size_t)M_ * D_ * 2));
  u16t*  f1bf   = (u16t*)(ws + alloc((size_t)M_ * DFF_ * 2));
  float* f2f    = (float*)(ws + alloc((size_t)M_ * D_ * 4));

  // 1) bf16 casts of activations + weights
  cast_bf16_kernel<<<1024, 256, 0, stream>>>(x,    xbf,    M_ * D_);
  cast_bf16_kernel<<<1024, 256, 0, stream>>>(Wqkv, wqkvbf, 3 * D_ * D_);
  cast_bf16_kernel<<<1024, 256, 0, stream>>>(Wo,   wobf,   D_ * D_);
  cast_bf16_kernel<<<1024, 256, 0, stream>>>(W1,   w1bf,   DFF_ * D_);
  cast_bf16_kernel<<<1024, 256, 0, stream>>>(W2,   w2bf,   D_ * DFF_);

  // 2) QKV projection: [M,1024] x [3072,1024]^T -> bf16 [M,3072]
  gemm_bf16_kernel<<<dim3(3 * D_ / 128, M_ / 128), 256, 0, stream>>>(
      xbf, D_, wqkvbf, D_, bqkv, nullptr, qkvbf, 3 * D_, D_, 0);

  // 3) Flash attention -> bf16 [M,1024]
  attn_kernel<<<dim3(B_ * H_, L_ / 16), 32, 0, stream>>>(qkvbf, obf);

  // 4) Output projection: [M,1024] x [1024,1024]^T -> f32 [M,1024]
  gemm_bf16_kernel<<<dim3(D_ / 128, M_ / 128), 256, 0, stream>>>(
      obf, D_, wobf, D_, bo, projf, nullptr, D_, D_, 0);

  // 5) LN1( x + proj ) -> hf (f32) + hbf (bf16)
  ln_kernel<<<M_, 256, 0, stream>>>(x, projf, ln1_a, ln1_b, hf, hbf);

  // 6) FFN1: relu([M,1024] x [4096,1024]^T + b1) -> bf16 [M,4096]
  gemm_bf16_kernel<<<dim3(DFF_ / 128, M_ / 128), 256, 0, stream>>>(
      hbf, D_, w1bf, D_, b1, nullptr, f1bf, DFF_, D_, 1);

  // 7) FFN2: [M,4096] x [1024,4096]^T + b2 -> f32 [M,1024]
  gemm_bf16_kernel<<<dim3(D_ / 128, M_ / 128), 256, 0, stream>>>(
      f1bf, DFF_, w2bf, DFF_, b2, f2f, nullptr, D_, DFF_, 0);

  // 8) LN2( h + ffn ) -> d_out (f32)
  ln_kernel<<<M_, 256, 0, stream>>>(hf, f2f, ln2_a, ln2_b, out, nullptr);
}
